// ExtractModel_83597243449626
// MI455X (gfx1250) — matrix operand order repaired and checked by reading, not compiled, not run on hardware
//
#include <hip/hip_runtime.h>
#include <hip/hip_bf16.h>

// ---------------------------------------------------------------------------
// MI455X (gfx1250) fused implementation.
// Roofline: 787MB mandatory f32 output writes (~34us @ 23.3TB/s) dominate;
// 88 GFLOP of GEMM work goes through v_wmma_f32_16x16x32_bf16 so compute is
// far under the memory time. Intermediates (ctx/ins bf16 reprs, 98MB) live in
// the workspace and stay L2-resident (192MB L2); the final output stream uses
// non-temporal b128 stores so it does not evict them.
// ---------------------------------------------------------------------------

typedef __bf16 bf16_t;
typedef __attribute__((ext_vector_type(16))) __bf16 v16bf;
typedef __attribute__((ext_vector_type(4)))  __bf16 v4bf;
typedef __attribute__((ext_vector_type(8)))  float  v8f;
typedef __attribute__((ext_vector_type(4)))  float  f4;

#define WMMA_BF16(a, b, c) \
    __builtin_amdgcn_wmma_f32_16x16x32_bf16(false, (a), false, (b), (short)0, (c), false, false)

constexpr int V_    = 16000;
constexpr int L_    = 12;
constexpr int K_    = 512;
constexpr int LOST_ = 512;
constexpr int D_    = 128;
constexpr int ROWS  = V_ * L_;          // 192000 flattened (v,l) rows
constexpr int KPATCH = 3 * D_;          // 384 conv patch K

// ---- WMMA fragment loaders (contiguous k = 16*h + j, same order in A and B) ------

// A fragment, 16x32 bf16: lane row M=lane&15, K=16*(lane>>4)+j, j=0..15.
// Source row-major bf16, stride ld.
__device__ __forceinline__ v16bf load_a_bf16(const bf16_t* base, int ld) {
    int lane = threadIdx.x & 31;
    int row  = lane & 15;
    int h    = lane >> 4;
    const bf16_t* p = base + row * ld + h * 16;
    v16bf a;
#pragma unroll
    for (int j = 0; j < 8; ++j) a[j] = p[j];
#pragma unroll
    for (int j = 0; j < 8; ++j) a[8 + j] = p[8 + j];
    return a;
}

// Same A fragment, but source is f32 (convert on the fly).
__device__ __forceinline__ v16bf load_a_f32(const float* base, int ld) {
    int lane = threadIdx.x & 31;
    int row  = lane & 15;
    int h    = lane >> 4;
    const float* p = base + row * ld + h * 16;
    v16bf a;
#pragma unroll
    for (int j = 0; j < 8; ++j) a[j] = (bf16_t)p[j];
#pragma unroll
    for (int j = 0; j < 8; ++j) a[8 + j] = (bf16_t)p[8 + j];
    return a;
}

// B fragment, 32x16 bf16, from a TRANSPOSED [N][K] row-major buffer so each
// lane reads 16 contiguous bf16 (32B): lane col N=lane&15, K=16*(lane>>4)+j.
__device__ __forceinline__ v16bf load_bt(const bf16_t* bt, int ldk) {
    int lane = threadIdx.x & 31;
    int col  = lane & 15;
    int kb   = (lane >> 4) * 16;
    const bf16_t* p = bt + col * ldk + kb;
    v16bf b;
#pragma unroll
    for (int j = 0; j < 16; ++j) b[j] = p[j];
    return b;
}

// B fragment from a [K][N] row-major f32 buffer (convert on the fly).
__device__ __forceinline__ v16bf load_b_f32_kn(const float* base, int ldn) {
    int lane = threadIdx.x & 31;
    int col  = lane & 15;
    int kb   = (lane >> 4) * 16;
    v16bf b;
#pragma unroll
    for (int j = 0; j < 16; ++j) b[j] = (bf16_t)base[(kb + j) * ldn + col];
    return b;
}

// ---------------------------------------------------------------------------
// Kernel 0c: relayout conv weights [Dout,Din,3] f32 -> [N=Dout][K=tap*128+din]
// bf16 (B^T layout for contiguous per-lane fragment loads).
// ---------------------------------------------------------------------------
__global__ __launch_bounds__(256) void k_conv_weights(const float* __restrict__ cw,
                                                      const float* __restrict__ iw,
                                                      bf16_t* __restrict__ wc,
                                                      bf16_t* __restrict__ wi) {
    int t = blockIdx.x * 256 + threadIdx.x;           // 2 * 128 * 384 = 98304 total
    int which = t / (D_ * KPATCH);
    int e     = t % (D_ * KPATCH);
    int n   = e / KPATCH;
    int k   = e % KPATCH;
    int tap = k >> 7;
    int din = k & 127;
    float v = (which ? iw : cw)[(n * D_ + din) * 3 + tap];
    (which ? wi : wc)[n * KPATCH + k] = (bf16_t)v;
}

// ---------------------------------------------------------------------------
// Kernel 0a: lost_unit_emb = aligner[512,512] @ known_unit_emb[512,128] -> bf16
// 256 tiles (32 M x 8 N); one 16x16 tile per wave, K loop = 16 chunks of 32.
// ---------------------------------------------------------------------------
__global__ __launch_bounds__(256) void k_lost(const float* __restrict__ aligner,
                                              const float* __restrict__ emb,
                                              bf16_t* __restrict__ lost_bf) {
    int w    = threadIdx.x >> 5;
    int tile = blockIdx.x * 8 + w;
    int m0 = (tile >> 3) * 16;
    int n0 = (tile & 7) * 16;
    v8f acc = {};
#pragma unroll
    for (int kc = 0; kc < 16; ++kc) {
        v16bf a = load_a_f32(aligner + m0 * K_ + kc * 32, K_);
        v16bf b = load_b_f32_kn(emb + (kc * 32) * D_ + n0, D_);
        acc = WMMA_BF16(a, b, acc);
    }
    int lane = threadIdx.x & 31;
    int col = lane & 15, ro = (lane >> 4) * 8;
#pragma unroll
    for (int r = 0; r < 8; ++r)
        lost_bf[(m0 + r + ro) * D_ + n0 + col] = (bf16_t)acc[r];
}

// ---------------------------------------------------------------------------
// Kernel 0b: unit_log_probs = log_softmax(emb @ lost^T), alignment = exp(lp).
// One 16-row tile per block (32 blocks); 8 waves x 4 N-tiles cover LOST=512.
// ---------------------------------------------------------------------------
__global__ __launch_bounds__(256) void k_unit_probs(const float* __restrict__ emb,
                                                    const bf16_t* __restrict__ lost_bf,
                                                    float* __restrict__ unit_lp,
                                                    float* __restrict__ align_out) {
    __shared__ float logits[16 * 512];
    __shared__ float red[256];
    __shared__ float rowstat[16];

    int tid  = threadIdx.x;
    int w    = tid >> 5;
    int lane = tid & 31;
    int m0   = blockIdx.x * 16;
    int col = lane & 15, ro = (lane >> 4) * 8;

    v16bf a[4];
#pragma unroll
    for (int kc = 0; kc < 4; ++kc)
        a[kc] = load_a_f32(emb + m0 * D_ + kc * 32, D_);

#pragma unroll
    for (int nt = 0; nt < 4; ++nt) {
        int n0 = w * 64 + nt * 16;
        v8f acc = {};
#pragma unroll
        for (int kc = 0; kc < 4; ++kc) {
            v16bf b = load_bt(lost_bf + n0 * D_ + kc * 32, D_);
            acc = WMMA_BF16(a[kc], b, acc);
        }
#pragma unroll
        for (int r = 0; r < 8; ++r)
            logits[(r + ro) * 512 + n0 + col] = acc[r];
    }
    __syncthreads();

    // row-wise LSE: 16 threads per row, 32 cols each, f4 LDS reads.
    int row = tid >> 4, s = tid & 15;
    float m = -1e30f;
#pragma unroll
    for (int j = 0; j < 8; ++j) {
        f4 x = *(const f4*)&logits[row * 512 + s * 32 + j * 4];
        m = fmaxf(m, fmaxf(fmaxf(x.x, x.y), fmaxf(x.z, x.w)));
    }
    red[tid] = m;
    __syncthreads();
    if (s == 0) {
        float mm = red[row * 16];
        for (int j = 1; j < 16; ++j) mm = fmaxf(mm, red[row * 16 + j]);
        rowstat[row] = mm;
    }
    __syncthreads();
    float mm = rowstat[row];
    float sum = 0.f;
#pragma unroll
    for (int j = 0; j < 8; ++j) {
        f4 x = *(const f4*)&logits[row * 512 + s * 32 + j * 4];
        sum += __expf(x.x - mm) + __expf(x.y - mm) + __expf(x.z - mm) + __expf(x.w - mm);
    }
    red[tid] = sum;
    __syncthreads();
    if (s == 0) {
        float ss = 0.f;
        for (int j = 0; j < 16; ++j) ss += red[row * 16 + j];
        rowstat[row] = mm + __logf(ss);
    }
    __syncthreads();

#pragma unroll
    for (int i = 0; i < 8; ++i) {              // 2048 float4 groups / 256 threads
        int e4 = tid + i * 256;
        int rr = e4 >> 7, n4 = (e4 & 127) << 2;
        f4 x = *(const f4*)&logits[rr * 512 + n4];
        float ls = rowstat[rr];
        f4 lp = {x.x - ls, x.y - ls, x.z - ls, x.w - ls};
        *(f4*)&unit_lp[(m0 + rr) * 512 + n4] = lp;     // re-read by k_scores: RT
        f4 ex = {__expf(lp.x), __expf(lp.y), __expf(lp.z), __expf(lp.w)};
        __builtin_nontemporal_store(ex, (f4*)&align_out[(size_t)(m0 + rr) * 512 + n4]);
    }
}

// ---------------------------------------------------------------------------
// Kernel 1: per-word conv (SAME, ksize 3) as GEMM [16(pad12) x 384]x[384x128],
// both conv paths. Conv weights live in 24 register-resident B fragments per
// wave (CDNA5 allows up to 1024 VGPRs/wave); emb patch staged in 4.6KB LDS.
// ---------------------------------------------------------------------------
__global__ __launch_bounds__(256) void k_conv(const float* __restrict__ emb_tab,
                                              const bf16_t* __restrict__ wc_bf,
                                              const bf16_t* __restrict__ wi_bf,
                                              const float* __restrict__ cb,
                                              const float* __restrict__ ib,
                                              const int* __restrict__ ids,
                                              const int* __restrict__ lens,
                                              bf16_t* __restrict__ ctx_ws,
                                              bf16_t* __restrict__ ins_ws) {
    __shared__ bf16_t emb_lds[18 * 128];   // row r holds emb[r-1]; rows 0,13..17 stay 0
    __shared__ int ids_sh[12];
    __shared__ int len_sh;

    int tid  = threadIdx.x;
    int w    = tid >> 5;
    int lane = tid & 31;
    int n0   = w * 16;                      // wave owns one of 8 Dout tiles
    int col = lane & 15, ro = (lane >> 4) * 8;

    float bias_c = cb[n0 + col];
    float bias_i = ib[n0 + col];

    // register-resident conv-weight B fragments: 12 K-chunks x 2 convs
    v16bf wcf[12], wif[12];
#pragma unroll
    for (int kc = 0; kc < 12; ++kc) {
        wcf[kc] = load_bt(wc_bf + n0 * KPATCH + kc * 32, KPATCH);
        wif[kc] = load_bt(wi_bf + n0 * KPATCH + kc * 32, KPATCH);
    }

    for (int e = tid; e < 18 * 128; e += 256) emb_lds[e] = (bf16_t)0.f;
    __syncthreads();

    for (int v = blockIdx.x; v < V_; v += gridDim.x) {
        if (tid < 12) ids_sh[tid] = ids[v * 12 + tid];
        if (tid == 0) len_sh = lens[v];
        __syncthreads();

        int len = len_sh;
        // gather 12x128 masked emb rows as f4 loads + packed v4bf LDS stores
        for (int g = tid; g < 12 * 32; g += 256) {
            int l  = g >> 5;
            int d4 = (g & 31) << 2;
            f4 val = {0.f, 0.f, 0.f, 0.f};
            if (l < len) val = *(const f4*)&emb_tab[ids_sh[l] * D_ + d4];
            v4bf h = {(bf16_t)val.x, (bf16_t)val.y, (bf16_t)val.z, (bf16_t)val.w};
            *(v4bf*)&emb_lds[(l + 1) * 128 + d4] = h;
        }
        __syncthreads();

        v8f acc_c = {}, acc_i = {};
#pragma unroll
        for (int kc = 0; kc < 12; ++kc) {        // K chunks never straddle a tap
            int tap   = kc >> 2;
            int dbase = (kc & 3) * 32;
            v16bf a = load_a_bf16(emb_lds + tap * 128 + dbase, 128);
            acc_c = WMMA_BF16(a, wcf[kc], acc_c);
            acc_i = WMMA_BF16(a, wif[kc], acc_i);
        }

#pragma unroll
        for (int r = 0; r < 8; ++r) {
            int row = r + ro;
            if (row < 12) {
                int o = (v * 12 + row) * D_ + n0 + col;
                ctx_ws[o] = (bf16_t)(acc_c[r] + bias_c);   // re-read by k_scores: RT
                ins_ws[o] = (bf16_t)(acc_i[r] + bias_i);
            }
        }
        // barriers at loop top separate next gather writes from these reads
    }
}

// ---------------------------------------------------------------------------
// Kernel 2: logits [16,512] = repr_tile @ lost^T per 16-row tile of the flat
// [192000,128] repr matrices; fused log-softmax + gather of unit_log_probs
// and NT-streamed b128 f32 output of `sub` and `ins`. lost B fragments are
// register resident (4 N-tiles x 4 K-chunks per wave = 128 VGPRs).
// ---------------------------------------------------------------------------
__global__ __launch_bounds__(256) void k_scores(const bf16_t* __restrict__ ctx_ws,
                                                const bf16_t* __restrict__ ins_ws,
                                                const bf16_t* __restrict__ lost_bf,
                                                const float* __restrict__ unit_lp,
                                                const int* __restrict__ ids_flat,
                                                float* __restrict__ sub_out,
                                                float* __restrict__ ins_out) {
    __shared__ float logits[16 * 512];
    __shared__ float red[256];
    __shared__ float rowstat[16];
    __shared__ int   rowid[16];

    int tid  = threadIdx.x;
    int w    = tid >> 5;
    int lane = tid & 31;
    int col = lane & 15, ro = (lane >> 4) * 8;

    v16bf bfrag[4][4];
#pragma unroll
    for (int nt = 0; nt < 4; ++nt)
#pragma unroll
        for (int kc = 0; kc < 4; ++kc)
            bfrag[nt][kc] = load_bt(lost_bf + (w * 64 + nt * 16) * D_ + kc * 32, D_);

    int row16 = tid >> 4, s = tid & 15;

    for (int tile = blockIdx.x; tile < ROWS / 16; tile += gridDim.x) {
        int m0 = tile * 16;
        if (tid < 16) rowid[tid] = ids_flat[m0 + tid];

        // prefetch next tile's A rows (4KB each) while this tile computes
        int ntile = tile + gridDim.x;
        if (ntile < ROWS / 16) {
            __builtin_prefetch((const char*)(ctx_ws + (size_t)ntile * 16 * D_) + tid * 16, 0, 1);
            __builtin_prefetch((const char*)(ins_ws + (size_t)ntile * 16 * D_) + tid * 16, 0, 1);
        }

#pragma unroll
        for (int phase = 0; phase < 2; ++phase) {
            const bf16_t* A = phase ? ins_ws : ctx_ws;

            v16bf a[4];
#pragma unroll
            for (int kc = 0; kc < 4; ++kc)
                a[kc] = load_a_bf16(A + m0 * D_ + kc * 32, D_);

            v8f acc[4] = {};
#pragma unroll
            for (int nt = 0; nt < 4; ++nt)
#pragma unroll
                for (int kc = 0; kc < 4; ++kc)
                    acc[nt] = WMMA_BF16(a[kc], bfrag[nt][kc], acc[nt]);

            __syncthreads();                      // prior readers of logits done
#pragma unroll
            for (int nt = 0; nt < 4; ++nt) {
                int n0 = w * 64 + nt * 16;
#pragma unroll
                for (int r = 0; r < 8; ++r)
                    logits[(r + ro) * 512 + n0 + col] = acc[nt][r];
            }
            __syncthreads();

            float m = -1e30f;
#pragma unroll
            for (int j = 0; j < 8; ++j) {
                f4 x = *(const f4*)&logits[row16 * 512 + s * 32 + j * 4];
                m = fmaxf(m, fmaxf(fmaxf(x.x, x.y), fmaxf(x.z, x.w)));
            }
            red[tid] = m;
            __syncthreads();
            if (s == 0) {
                float mm = red[row16 * 16];
                for (int j = 1; j < 16; ++j) mm = fmaxf(mm, red[row16 * 16 + j]);
                rowstat[row16] = mm;
            }
            __syncthreads();
            float mm = rowstat[row16];
            float sum = 0.f;
#pragma unroll
            for (int j = 0; j < 8; ++j) {
                f4 x = *(const f4*)&logits[row16 * 512 + s * 32 + j * 4];
                sum += __expf(x.x - mm) + __expf(x.y - mm) +
                       __expf(x.z - mm) + __expf(x.w - mm);
            }
            red[tid] = sum;
            __syncthreads();
            if (s == 0) {
                float ss = 0.f;
                for (int j = 0; j < 16; ++j) ss += red[row16 * 16 + j];
                rowstat[row16] = mm + __logf(ss);
            }
            __syncthreads();

            // NT b128 output stream (dominant traffic; bypass L2 retention)
            if (phase == 0) {
#pragma unroll
                for (int i = 0; i < 8; ++i) {
                    int e4 = tid + i * 256;
                    int rr = e4 >> 7, n4 = (e4 & 127) << 2;
                    f4 x = *(const f4*)&logits[rr * 512 + n4];
                    float ls = rowstat[rr];
                    f4 g = *(const f4*)&unit_lp[rowid[rr] * 512 + n4];
                    f4 o = {-(0.5f * (x.x - ls) + 0.5f * g.x),
                            -(0.5f * (x.y - ls) + 0.5f * g.y),
                            -(0.5f * (x.z - ls) + 0.5f * g.z),
                            -(0.5f * (x.w - ls) + 0.5f * g.w)};
                    __builtin_nontemporal_store(o, (f4*)&sub_out[(size_t)(m0 + rr) * 512 + n4]);
                }
            } else {
#pragma unroll
                for (int i = 0; i < 8; ++i) {
                    int e4 = tid + i * 256;
                    int rr = e4 >> 7, n4 = (e4 & 127) << 2;
                    f4 x = *(const f4*)&logits[rr * 512 + n4];
                    float ls = rowstat[rr];
                    f4 o = {3.5f - 0.5f * (x.x - ls), 3.5f - 0.5f * (x.y - ls),
                            3.5f - 0.5f * (x.z - ls), 3.5f - 0.5f * (x.w - ls)};
                    __builtin_nontemporal_store(o, (f4*)&ins_out[(size_t)(m0 + rr) * 512 + n4]);
                }
            }
        }
    }
}

// ---------------------------------------------------------------------------

extern "C" void kernel_launch(void* const* d_in, const int* in_sizes, int n_in,
                              void* d_out, int out_size, void* d_ws, size_t ws_size,
                              hipStream_t stream) {
    (void)in_sizes; (void)n_in; (void)out_size; (void)ws_size;

    const float* emb     = (const float*)d_in[0];   // [512,128]
    const float* aligner = (const float*)d_in[1];   // [512,512]
    const float* conv_w  = (const float*)d_in[2];   // [128,128,3]
    const float* conv_b  = (const float*)d_in[3];   // [128]
    const float* ins_w   = (const float*)d_in[4];   // [128,128,3]
    const float* ins_b   = (const float*)d_in[5];   // [128]
    const int*   ids     = (const int*)d_in[6];     // [16000,12]
    const int*   lens    = (const int*)d_in[7];     // [16000]

    char*  ws  = (char*)d_ws;
    size_t off = 0;
    auto alloc = [&](size_t bytes) -> void* {
        void* p = ws + off;
        off += (bytes + 255) & ~(size_t)255;
        return p;
    };
    bf16_t* lost_bf = (bf16_t*)alloc((size_t)LOST_ * D_ * sizeof(bf16_t));   // 128KB
    bf16_t* wc_bf   = (bf16_t*)alloc((size_t)D_ * KPATCH * sizeof(bf16_t)); // 96KB
    bf16_t* wi_bf   = (bf16_t*)alloc((size_t)D_ * KPATCH * sizeof(bf16_t)); // 96KB
    float*  unit_lp = (float*)alloc((size_t)K_ * LOST_ * sizeof(float));     // 1MB
    bf16_t* ctx_ws  = (bf16_t*)alloc((size_t)ROWS * D_ * sizeof(bf16_t));    // 49MB
    bf16_t* ins_ws  = (bf16_t*)alloc((size_t)ROWS * D_ * sizeof(bf16_t));    // 49MB

    float* sub_out   = (float*)d_out;
    float* ins_out   = sub_out + (size_t)ROWS * LOST_;
    float* align_out = ins_out + (size_t)ROWS * LOST_;

    k_conv_weights<<<(2 * D_ * KPATCH) / 256, 256, 0, stream>>>(conv_w, ins_w, wc_bf, wi_bf);
    k_lost<<<32, 256, 0, stream>>>(aligner, emb, lost_bf);
    k_unit_probs<<<32, 256, 0, stream>>>(emb, lost_bf, unit_lp, align_out);
    k_conv<<<1024, 256, 0, stream>>>(emb, wc_bf, wi_bf, conv_b, ins_b, ids, lens, ctx_ws, ins_ws);
    k_scores<<<2048, 256, 0, stream>>>(ctx_ws, ins_ws, lost_bf, unit_lp, ids, sub_out, ins_out);
}
